// DRAWModel_64828236366303
// MI455X (gfx1250) — matrix operand order repaired and checked by reading, not compiled, not run on hardware
//
#include <hip/hip_runtime.h>

typedef __attribute__((ext_vector_type(8)))  __bf16 bf16x8;
typedef __attribute__((ext_vector_type(16))) __bf16 bf16x16;
typedef __attribute__((ext_vector_type(8)))  float  f32x8;

#define IMG        64
#define BATCH      512
#define UNITS      256
#define PATCH_DIM  9216          // 32*32*9
#define ENC_K      9984          // patches(9216) + h_dec(256) + h_dec(256) + h_enc(256)
#define DEC_K      512           // z(256) + h_dec(256)
#define GATES      1024

__device__ __forceinline__ float sigmoidf_(float x) { return 1.0f / (1.0f + __expf(-x)); }

// Async global->LDS 16B copy (GLOBAL_LOAD_ASYNC_TO_LDS_B128, ASYNCcnt).
// lds_addr: per-lane 32-bit LDS byte address (low 32 bits of generic pointer,
// per ISA 10.2 aperture rules); g: per-lane global source address.
__device__ __forceinline__ void async_ld_b128(unsigned lds_addr, const void* g) {
    asm volatile("global_load_async_to_lds_b128 %0, %1, off"
                 :: "v"(lds_addr), "v"((unsigned long long)(uintptr_t)g)
                 : "memory");
}
__device__ __forceinline__ void wait_async0() {
#if __has_builtin(__builtin_amdgcn_s_wait_asynccnt)
    __builtin_amdgcn_s_wait_asynccnt(0);
#else
    asm volatile("s_wait_asynccnt 0x0" ::: "memory");
#endif
}

// ---------------------------------------------------------------------------
// fp32 [Krows, N] -> bf16 transposed into out[N][ldout] (caller offsets `out`
// for K-concatenated weights). 32x32 LDS tile; both sides coalesced.
// ---------------------------------------------------------------------------
__global__ void cvt_transpose_bf16(const float* __restrict__ in, __bf16* __restrict__ out,
                                   int ldout, int Ndim) {
    __shared__ float tile[32][33];
    const int kt = blockIdx.y * 32, nt = blockIdx.x * 32;
    const int tx = threadIdx.x & 31, ty = threadIdx.x >> 5;   // 32 x 8
#pragma unroll
    for (int j = 0; j < 32; j += 8)
        tile[ty + j][tx] = in[(size_t)(kt + ty + j) * Ndim + nt + tx];
    __syncthreads();
#pragma unroll
    for (int j = 0; j < 32; j += 8)
        out[(size_t)(nt + ty + j) * ldout + kt + tx] = (__bf16)tile[tx][ty + j];
}

// ---------------------------------------------------------------------------
// Build encoder input: x_hat = x - sigmoid(canvas); 3x3 stride-2 SAME patches
// (TF layout [h', w', 9]) -> bf16 into enc_in[:, 0:9216].
// ---------------------------------------------------------------------------
__global__ void build_encin(const float* __restrict__ x, const float* __restrict__ canvas,
                            __bf16* __restrict__ enc_in) {
    __shared__ float xhat[IMG * IMG];
    const int b = blockIdx.x, t = threadIdx.x;
    const float* xb = x + (size_t)b * IMG * IMG;
    const float* cb = canvas + (size_t)b * IMG * IMG;
#pragma unroll
    for (int i = 0; i < 16; ++i) {
        int p = t + i * 256;
        xhat[p] = xb[p] - sigmoidf_(cb[p]);
    }
    __syncthreads();
    __bf16* eb = enc_in + (size_t)b * ENC_K;
    for (int i = 0; i < 36; ++i) {            // 9216 / 256
        int e = t + i * 256;
        int pe = e % 9, cell = e / 9;
        int kw = pe % 3, kh = pe / 3;
        int j = cell & 31, ii = cell >> 5;
        int r = 2 * ii + kh, c = 2 * j + kw;  // SAME: pad_before=0, pad_after=1
        float v = (r < IMG && c < IMG) ? xhat[r * IMG + c] : 0.0f;
        eb[e] = (__bf16)v;
    }
}

// ---------------------------------------------------------------------------
// bf16 GEMM via v_wmma_f32_16x16x32_bf16 with pre-transposed weights.
//   C[M,N] (fp32) = A[M,K](bf16) @ W(= WT[N,K] in memory) (+ bias) (+= if acc)
// Block: 256 threads (8 waves), 128(M) x 128(N) tile. Wave: 64x32 -> 8 v8f acc.
// Tiles ([128 rows x 32 K] for both A and WT) are fetched with async
// global->LDS B128 loads into a 2-stage LDS double buffer: issue chunk k+1
// into buf^1, compute chunk k from buf, then s_wait_asynccnt 0 + one barrier
// (proves own data landed AND all waves finished reading the buffer that
// iteration k+1 will overwrite). M,N multiples of 128; K multiple of 32.
// ---------------------------------------------------------------------------
__global__ void gemm_bf16(const __bf16* __restrict__ A, int lda,
                          const __bf16* __restrict__ WT, int ldwt,
                          float* __restrict__ C, int ldc,
                          int K, const float* __restrict__ bias, int accumulate) {
    __shared__ __align__(16) __bf16 Alds[2][128][40];   // 80B pitch -> 16B aligned chunks
    __shared__ __align__(16) __bf16 Blds[2][128][40];   // rows indexed by N (WT rows)

    const int t = threadIdx.x;
    const int lane = t & 31, wave = t >> 5;
    const int waveM = wave & 1, waveN = wave >> 1;   // 2 waves M x 4 waves N
    const int l16 = lane & 15, half = (lane >> 4) & 1;
    const int mblk = blockIdx.y * 128;
    const int nblk = blockIdx.x * 128;

    // Tile-fetch coordinates (identical for A and WT): 128x32 bf16; each
    // thread owns rows r0 and r0+64 at column c0 (one 16B async load each).
    const int r0 = t >> 2;              // 0..63
    const int c0 = (t & 3) * 8;         // 0,8,16,24
    const __bf16* Aptr = A + (size_t)(mblk + r0) * lda + c0;
    const __bf16* Bptr = WT + (size_t)(nblk + r0) * ldwt + c0;

    const unsigned la = (unsigned)(uintptr_t)&Alds[0][r0][c0];
    const unsigned lb = (unsigned)(uintptr_t)&Blds[0][r0][c0];
    const unsigned rowHop = 64u * 80u;          // 64 rows * 80B pitch
    const unsigned bufHop = 128u * 80u;         // buffer stride (10240B)

    f32x8 acc[4][2] = {};
    const int nk = K / 32;

    // prologue: async-fetch chunk 0 into buffer 0
    async_ld_b128(la,          Aptr);
    async_ld_b128(la + rowHop, Aptr + (size_t)64 * lda);
    async_ld_b128(lb,          Bptr);
    async_ld_b128(lb + rowHop, Bptr + (size_t)64 * ldwt);
    wait_async0();
    __syncthreads();

    for (int i = 0; i < nk; ++i) {
        const int buf = i & 1;
        // issue async loads for chunk i+1 into the other buffer
        if (i + 1 < nk) {
            const int kn = (i + 1) * 32;
            const unsigned bo = (unsigned)(buf ^ 1) * bufHop;
            async_ld_b128(la + bo,          Aptr + kn);
            async_ld_b128(la + bo + rowHop, Aptr + (size_t)64 * lda + kn);
            async_ld_b128(lb + bo,          Bptr + kn);
            async_ld_b128(lb + bo + rowHop, Bptr + (size_t)64 * ldwt + kn);
        }
        if (i + 2 < nk) {   // global_prefetch_b8 two chunks ahead
            __builtin_prefetch(Aptr + (i + 2) * 32, 0, 1);
            __builtin_prefetch(Bptr + (i + 2) * 32, 0, 1);
        }

        // A fragment (16x32): lane<16 -> K {0..7,16..23}; lane>=16 -> K {8..15,24..31}
        union Frag { bf16x16 v; bf16x8 h[2]; };
        Frag afrag[4], bfrag[2];
#pragma unroll
        for (int mf = 0; mf < 4; ++mf) {
            int row = waveM * 64 + mf * 16 + l16;
            afrag[mf].h[0] = *(const bf16x8*)&Alds[buf][row][8 * half];
            afrag[mf].h[1] = *(const bf16x8*)&Alds[buf][row][16 + 8 * half];
        }
        // B fragment (32x16): col = lane&15, lane<16 -> K {0..7,16..23}; lane>=16 -> K {8..15,24..31}
#pragma unroll
        for (int nf = 0; nf < 2; ++nf) {
            int col = waveN * 32 + nf * 16 + l16;
            bfrag[nf].h[0] = *(const bf16x8*)&Blds[buf][col][8 * half];
            bfrag[nf].h[1] = *(const bf16x8*)&Blds[buf][col][16 + 8 * half];
        }
#pragma unroll
        for (int mf = 0; mf < 4; ++mf)
#pragma unroll
            for (int nf = 0; nf < 2; ++nf)
                acc[mf][nf] = __builtin_amdgcn_wmma_f32_16x16x32_bf16(
                    false, afrag[mf].v, false, bfrag[nf].v,
                    (short)0, acc[mf][nf], false, false);

        if (i + 1 < nk) {
            wait_async0();      // own async loads into buf^1 landed
            __syncthreads();    // everyone's landed; everyone done reading buf
        }
    }

    // D layout: VGPR r -> M = r + 8*half, N = lane&15
#pragma unroll
    for (int mf = 0; mf < 4; ++mf) {
#pragma unroll
        for (int nf = 0; nf < 2; ++nf) {
            int col = nblk + waveN * 32 + nf * 16 + l16;
            float bv = bias ? bias[col] : 0.0f;
#pragma unroll
            for (int r = 0; r < 8; ++r) {
                int row = mblk + waveM * 64 + mf * 16 + r + 8 * half;
                float v = acc[mf][nf][r] + bv;
                float* p = C + (size_t)row * ldc + col;
                if (accumulate) *p += v; else *p = v;
            }
        }
    }
}

// ---------------------------------------------------------------------------
// LSTM pointwise: gates [512,1024] (i,f,g,o), c in/out. Writes bf16 copies of
// h directly into the input buffers of downstream GEMMs (up to 4 strided dsts).
// ---------------------------------------------------------------------------
__global__ void lstm_point(const float* __restrict__ gates, float* __restrict__ c,
                           float* __restrict__ h_out,
                           __bf16* d0, int ld0, __bf16* d1, int ld1,
                           __bf16* d2, int ld2, __bf16* d3, int ld3) {
    int idx = blockIdx.x * blockDim.x + threadIdx.x;   // 512*256 threads
    int b = idx >> 8, u = idx & 255;
    const float* g = gates + (size_t)b * GATES;
    float i_ = sigmoidf_(g[u]);
    float f_ = sigmoidf_(g[256 + u]);
    float gg = tanhf(g[512 + u]);
    float o_ = sigmoidf_(g[768 + u]);
    float cn = f_ * c[idx] + i_ * gg;
    float hn = o_ * tanhf(cn);
    c[idx] = cn;
    if (h_out) h_out[idx] = hn;
    __bf16 hb = (__bf16)hn;
    if (d0) d0[(size_t)b * ld0 + u] = hb;
    if (d1) d1[(size_t)b * ld1 + u] = hb;
    if (d2) d2[(size_t)b * ld2 + u] = hb;
    if (d3) d3[(size_t)b * ld3 + u] = hb;
}

// ---------------------------------------------------------------------------
// Attention: logits = h_enc @ W_enc + b_enc; softmax over 10; z = attn @ W_enc^T
// One block per batch row; writes z in bf16 into dec_in[:, 0:256].
// ---------------------------------------------------------------------------
__global__ void attn_z(const float* __restrict__ h_enc, const float* __restrict__ W_enc,
                       const float* __restrict__ b_enc, __bf16* __restrict__ dec_in) {
    __shared__ float hs[UNITS];
    __shared__ float logit[10];
    __shared__ float attn[10];
    const int b = blockIdx.x, t = threadIdx.x;
    hs[t] = h_enc[(size_t)b * UNITS + t];
    __syncthreads();
    if (t < 10) {
        float s = b_enc[t];
        for (int u = 0; u < UNITS; ++u) s += hs[u] * W_enc[u * 10 + t];
        logit[t] = s;
    }
    __syncthreads();
    if (t < 10) {
        float m = logit[0];
#pragma unroll
        for (int k = 1; k < 10; ++k) m = fmaxf(m, logit[k]);
        float den = 0.0f;
#pragma unroll
        for (int k = 0; k < 10; ++k) den += __expf(logit[k] - m);
        attn[t] = __expf(logit[t] - m) / den;
    }
    __syncthreads();
    float z = 0.0f;
#pragma unroll
    for (int k = 0; k < 10; ++k) z += attn[k] * W_enc[t * 10 + k];
    dec_in[(size_t)b * DEC_K + t] = (__bf16)z;
}

// ---------------------------------------------------------------------------
extern "C" void kernel_launch(void* const* d_in, const int* in_sizes, int n_in,
                              void* d_out, int out_size, void* d_ws, size_t ws_size,
                              hipStream_t stream) {
    (void)in_sizes; (void)n_in; (void)out_size; (void)ws_size;
    const float* x          = (const float*)d_in[0];
    const float* enc_kernel = (const float*)d_in[1];   // [9728, 1024]
    const float* enc_rec    = (const float*)d_in[2];   // [256, 1024]
    const float* enc_bias   = (const float*)d_in[3];   // [1024]
    const float* dec_kernel = (const float*)d_in[4];   // [256, 1024]
    const float* dec_rec    = (const float*)d_in[5];   // [256, 1024]
    const float* dec_bias   = (const float*)d_in[6];   // [1024]
    const float* W_enc      = (const float*)d_in[7];   // [256, 10]
    const float* b_enc      = (const float*)d_in[8];   // [10]
    const float* W_dec      = (const float*)d_in[9];   // [256, 4096]
    const float* b_dec      = (const float*)d_in[10];  // [4096]
    float* canvas = (float*)d_out;                     // [512, 4096]

    // Workspace carve-out (~38 MB)
    char* ws = (char*)d_ws;
    size_t off = 0;
    auto carve = [&](size_t bytes) -> char* {
        char* p = ws + off;
        off = (off + bytes + 255) & ~(size_t)255;
        return p;
    };
    // Transposed bf16 weights: WT[N][K]
    __bf16* encT   = (__bf16*)carve((size_t)GATES * ENC_K * 2);  // [1024][9984]
    __bf16* decT   = (__bf16*)carve((size_t)GATES * DEC_K * 2);  // [1024][512]
    __bf16* wdecT  = (__bf16*)carve((size_t)4096 * UNITS * 2);   // [4096][256]
    __bf16* enc_in = (__bf16*)carve((size_t)BATCH * ENC_K * 2);  // [patches|h_dec|h_dec|h_enc]
    __bf16* dec_in = (__bf16*)carve((size_t)BATCH * DEC_K * 2);  // [z|h_dec]
    __bf16* hdec16 = (__bf16*)carve((size_t)BATCH * UNITS * 2);
    float*  gates  = (float*)carve((size_t)BATCH * GATES * 4);
    float*  h_enc  = (float*)carve((size_t)BATCH * UNITS * 4);
    float*  c_enc  = (float*)carve((size_t)BATCH * UNITS * 4);
    float*  c_dec  = (float*)carve((size_t)BATCH * UNITS * 4);

    // Zero recurrent state + canvas + input buffers (capture-safe)
    hipMemsetAsync(canvas, 0, (size_t)BATCH * 4096 * 4, stream);
    hipMemsetAsync(enc_in, 0, (size_t)BATCH * ENC_K * 2, stream);
    hipMemsetAsync(dec_in, 0, (size_t)BATCH * DEC_K * 2, stream);
    hipMemsetAsync(h_enc, 0, (size_t)BATCH * UNITS * 4, stream);
    hipMemsetAsync(c_enc, 0, (size_t)BATCH * UNITS * 4, stream);
    hipMemsetAsync(c_dec, 0, (size_t)BATCH * UNITS * 4, stream);

    // One-time weight convert+transpose to bf16 WT[N][K] (K-concat: kernel;rec)
    cvt_transpose_bf16<<<dim3(GATES / 32, 9728 / 32), 256, 0, stream>>>(
        enc_kernel, encT, ENC_K, GATES);
    cvt_transpose_bf16<<<dim3(GATES / 32, 256 / 32), 256, 0, stream>>>(
        enc_rec, encT + 9728, ENC_K, GATES);
    cvt_transpose_bf16<<<dim3(GATES / 32, 256 / 32), 256, 0, stream>>>(
        dec_kernel, decT, DEC_K, GATES);
    cvt_transpose_bf16<<<dim3(GATES / 32, 256 / 32), 256, 0, stream>>>(
        dec_rec, decT + 256, DEC_K, GATES);
    cvt_transpose_bf16<<<dim3(4096 / 32, 256 / 32), 256, 0, stream>>>(
        W_dec, wdecT, UNITS, 4096);

    for (int s = 0; s < 10; ++s) {
        build_encin<<<BATCH, 256, 0, stream>>>(x, canvas, enc_in);

        // Encoder: [512, 9984] @ [9984, 1024] + enc_bias
        gemm_bf16<<<dim3(8, 4), 256, 0, stream>>>(enc_in, ENC_K, encT, ENC_K,
                                                  gates, GATES, ENC_K, enc_bias, 0);
        // h_enc (fp32 for attention) + bf16 into enc_in[:, 9728:9984]
        lstm_point<<<BATCH, 256, 0, stream>>>(gates, c_enc, h_enc,
                                              enc_in + 9728, ENC_K,
                                              nullptr, 0, nullptr, 0, nullptr, 0);
        attn_z<<<BATCH, 256, 0, stream>>>(h_enc, W_enc, b_enc, dec_in);

        // Decoder: [512, 512] @ [512, 1024] + dec_bias
        gemm_bf16<<<dim3(8, 4), 256, 0, stream>>>(dec_in, DEC_K, decT, DEC_K,
                                                  gates, GATES, DEC_K, dec_bias, 0);
        // h_dec bf16 fan-out: enc_in[:,9216:9472], enc_in[:,9472:9728], dec_in[:,256:512], hdec16
        lstm_point<<<BATCH, 256, 0, stream>>>(gates, c_dec, nullptr,
                                              enc_in + 9216, ENC_K,
                                              enc_in + 9472, ENC_K,
                                              dec_in + 256, DEC_K,
                                              hdec16, UNITS);
        // canvas += [512,256] @ [256,4096] + b_dec
        gemm_bf16<<<dim3(32, 4), 256, 0, stream>>>(hdec16, UNITS, wdecT, UNITS,
                                                   canvas, 4096, UNITS, b_dec, 1);
    }
}
